// BasicCNN_LSTM_12360915878065
// MI455X (gfx1250) — matrix operand order repaired and checked by reading, not compiled, not run on hardware
//
#include <hip/hip_runtime.h>
#include <hip/hip_bf16.h>

typedef __attribute__((ext_vector_type(16))) _Float16 v16h;
typedef __attribute__((ext_vector_type(8)))  float    v8f;

union ABfrag { v16h v; uint32_t u[8]; };

// ---------------- workspace layout ----------------
// act buffers: [1024][30][30][48] f16 (1-pixel zero halo around 28x28)
constexpr size_t ACT_BYTES  = (size_t)1024 * 30 * 30 * 48 * 2;   // 88,473,600
constexpr size_t OFF_ACT1   = 0;
constexpr size_t OFF_ACT2   = ACT_BYTES;
constexpr size_t OFF_FEAT   = 2 * ACT_BYTES;                     // featsum [1024][48] f32
constexpr size_t FEAT_BYTES = (size_t)1024 * 48 * 4;             // 196,608
constexpr size_t OFF_BT2    = OFF_FEAT + FEAT_BYTES;             // Bt2 [48][448] f16
constexpr size_t BT_BYTES   = (size_t)48 * 448 * 2;              // 43,008
constexpr size_t OFF_BT3    = OFF_BT2 + BT_BYTES;
constexpr size_t WS_NEEDED  = OFF_BT3 + BT_BYTES;
constexpr size_t ZERO_BYTES = OFF_FEAT + FEAT_BYTES;             // act1+act2+featsum

// ---------------- zero workspace ----------------
__global__ void zero_ws_kernel(uint4* __restrict__ p, int n16) {
    uint4 z; z.x = z.y = z.z = z.w = 0u;
    for (int i = blockIdx.x * blockDim.x + threadIdx.x; i < n16;
         i += gridDim.x * blockDim.x)
        p[i] = z;
}

// ---------------- weight prep: w[3][3][48][48] f32 -> Bt[48][448] f16 (k = tap*48 + ci, zero-padded) ----------------
__global__ void prep_weights_kernel(const float* __restrict__ w2,
                                    const float* __restrict__ w3,
                                    _Float16* __restrict__ Bt2,
                                    _Float16* __restrict__ Bt3) {
    int i = blockIdx.x * blockDim.x + threadIdx.x;
    if (i >= 48 * 448) return;
    int co = i / 448;
    int k  = i - co * 448;
    _Float16 v2 = (_Float16)0.f, v3 = (_Float16)0.f;
    if (k < 432) {                       // w flat index: (tap*48 + ci)*48 + co
        v2 = (_Float16)w2[k * 48 + co];
        v3 = (_Float16)w3[k * 48 + co];
    }
    Bt2[co * 448 + k] = v2;
    Bt3[co * 448 + k] = v3;
}

// ---------------- conv1: C=1 -> 48, relu, write f16 into halo layout ----------------
__global__ __launch_bounds__(256)
void conv1_kernel(const float* __restrict__ x, const float* __restrict__ w1,
                  const float* __restrict__ b1, _Float16* __restrict__ act1) {
    int p = blockIdx.x * 256 + threadIdx.x;       // 1024*784 pixels, exact grid
    int frame = p / 784;
    int pix   = p - frame * 784;
    int y = pix / 28, xx = pix - (pix / 28) * 28;
    float in[9];
#pragma unroll
    for (int ky = 0; ky < 3; ++ky)
#pragma unroll
        for (int kx = 0; kx < 3; ++kx) {
            int yy = y + ky - 1, xv = xx + kx - 1;
            bool ok = (yy >= 0) & (yy < 28) & (xv >= 0) & (xv < 28);
            in[ky * 3 + kx] = ok ? x[(frame * 28 + yy) * 28 + xv] : 0.f;
        }
    int ob = ((frame * 30 + y + 1) * 30 + (xx + 1)) * 48;
#pragma unroll 4
    for (int co = 0; co < 48; ++co) {
        float a = b1[co];
#pragma unroll
        for (int t = 0; t < 9; ++t) a += in[t] * w1[t * 48 + co];
        act1[ob + co] = (_Float16)fmaxf(a, 0.f);
    }
}

// ---------------- conv 48->48 via implicit-GEMM WMMA f16 (K=432 padded to 448) ----------------
// act_in: [1024][30][30][48] f16 halo layout; Bt: [48][448] f16 transposed weights.
// POOL=false: relu -> act_out. POOL=true: relu -> atomicAdd channel sums into featsum[1024][48].
template <bool POOL>
__global__ __launch_bounds__(256)
void conv48_wmma_kernel(const _Float16* __restrict__ act_in,
                        const _Float16* __restrict__ Bt,
                        const float* __restrict__ bias,
                        _Float16* __restrict__ act_out,
                        float* __restrict__ featsum) {
    const int lane  = threadIdx.x & 31;
    const int wid   = threadIdx.x >> 5;
    const int mtile = blockIdx.x * 8 + wid;       // 50176 tiles, exact grid
    const int frame = mtile / 49;
    const int tif   = mtile - frame * 49;         // 16-pixel tile within frame
    const int col   = lane & 15;                  // N within tile / M row for A
    const int half  = lane >> 4;
    const int khalf = half * 8;

    // A-fragment row pixel (same M rows in both lane halves)
    const int pix = tif * 16 + col;
    const int ay = pix / 28, ax = pix - (pix / 28) * 28;
    int rowoff[9];
#pragma unroll
    for (int ky = 0; ky < 3; ++ky)
#pragma unroll
        for (int kx = 0; kx < 3; ++kx)
            rowoff[ky * 3 + kx] = ((frame * 30 + ay + ky) * 30 + (ax + kx)) * 48;

    v8f acc0, acc1, acc2;
    {
        float c0 = bias[col], c1 = bias[col + 16], c2 = bias[col + 32];
#pragma unroll
        for (int r = 0; r < 8; ++r) { acc0[r] = c0; acc1[r] = c1; acc2[r] = c2; }
    }

    const uint4* bt0 = (const uint4*)(Bt + (size_t)(col)      * 448);
    const uint4* bt1 = (const uint4*)(Bt + (size_t)(col + 16) * 448);
    const uint4* bt2 = (const uint4*)(Bt + (size_t)(col + 32) * 448);

#pragma unroll
    for (int kk = 0; kk < 14; ++kk) {
        // ---- A fragment per CDNA5 16-bit A layout: elem e -> K = (e<8?e:e+8) + 8*half ----
        ABfrag a;
#pragma unroll
        for (int j = 0; j < 8; ++j) {
            const int pairbase = (2 * j < 8) ? 2 * j : 2 * j + 8;  // compile-time
            const int kmin = 32 * kk + pairbase;
            if (kmin + 8 < 432) {                                   // compile-time mask
                const int tlo = kmin / 48;
                const int thi = (kmin + 8) / 48;
                int ro, ci;
                if (tlo == thi) {
                    ro = rowoff[tlo];
                    ci = kmin - tlo * 48 + khalf;
                } else {  // lane halves straddle a conv tap: select precomputed row ptr
                    ro = half ? rowoff[thi] : rowoff[tlo];
                    ci = half ? (kmin + 8 - thi * 48) : (kmin - tlo * 48);
                }
                a.u[j] = *(const uint32_t*)(act_in + ro + ci);
            } else {
                a.u[j] = 0u;
            }
        }
        // ---- B fragments: lane holds K 8*half..+7 and 16+8*half..+7 of its column -> 2x b128 ----
        const int bo = 4 * kk + half;  // uint4 index: (32*kk + 8*half)*2B / 16B
        ABfrag b0, b1, b2;
        *(uint4*)&b0.u[0] = bt0[bo]; *(uint4*)&b0.u[4] = bt0[bo + 2];
        *(uint4*)&b1.u[0] = bt1[bo]; *(uint4*)&b1.u[4] = bt1[bo + 2];
        *(uint4*)&b2.u[0] = bt2[bo]; *(uint4*)&b2.u[4] = bt2[bo + 2];

        acc0 = __builtin_amdgcn_wmma_f32_16x16x32_f16(false, a.v, false, b0.v,
                                                      (short)0, acc0, false, false);
        acc1 = __builtin_amdgcn_wmma_f32_16x16x32_f16(false, a.v, false, b1.v,
                                                      (short)0, acc1, false, false);
        acc2 = __builtin_amdgcn_wmma_f32_16x16x32_f16(false, a.v, false, b2.v,
                                                      (short)0, acc2, false, false);
    }

    if (!POOL) {
        // D elem r -> pixel m = r + 8*half, channel col (+16*ntile); relu + f16 store into halo layout
#pragma unroll
        for (int r = 0; r < 8; ++r) {
            int po = tif * 16 + r + 8 * half;
            int oy = po / 28, ox = po - (po / 28) * 28;
            int ob = ((frame * 30 + oy + 1) * 30 + (ox + 1)) * 48;
            act_out[ob + col]      = (_Float16)fmaxf(acc0[r], 0.f);
            act_out[ob + col + 16] = (_Float16)fmaxf(acc1[r], 0.f);
            act_out[ob + col + 32] = (_Float16)fmaxf(acc2[r], 0.f);
        }
    } else {
        float s0 = 0.f, s1 = 0.f, s2 = 0.f;
#pragma unroll
        for (int r = 0; r < 8; ++r) {
            s0 += fmaxf(acc0[r], 0.f);
            s1 += fmaxf(acc1[r], 0.f);
            s2 += fmaxf(acc2[r], 0.f);
        }
        atomicAdd(&featsum[frame * 48 + col],      s0);
        atomicAdd(&featsum[frame * 48 + col + 16], s1);
        atomicAdd(&featsum[frame * 48 + col + 32], s2);
    }
}

// ---------------- LSTM (B=32,U=8,T=32) + BN2 + head; BN1 folded into feature read ----------------
__global__ __launch_bounds__(256)
void lstm_kernel(const float* __restrict__ featsum,
                 const float* __restrict__ bn1_g, const float* __restrict__ bn1_b,
                 const float* __restrict__ bn1_m, const float* __restrict__ bn1_v,
                 const float* __restrict__ wf,  const float* __restrict__ bf,
                 const float* __restrict__ wi1, const float* __restrict__ bi1,
                 const float* __restrict__ wi2, const float* __restrict__ bi2,
                 const float* __restrict__ wo,  const float* __restrict__ bo,
                 const float* __restrict__ bn2_g, const float* __restrict__ bn2_b,
                 const float* __restrict__ bn2_m, const float* __restrict__ bn2_v,
                 const float* __restrict__ w_out, const float* __restrict__ b_out,
                 float* __restrict__ out) {
    __shared__ float WL[4][56][8];
    __shared__ float BLs[4][8];
    __shared__ float sL[48], tL[48];
    __shared__ float featL[32][48];
    __shared__ float hL[32][8];

    const int tid = threadIdx.x;
    for (int i = tid; i < 448; i += 256) {
        int j = i / 8, u = i - (i / 8) * 8;
        WL[0][j][u] = wf[i];  WL[1][j][u] = wi1[i];
        WL[2][j][u] = wi2[i]; WL[3][j][u] = wo[i];
    }
    if (tid < 8) {
        BLs[0][tid] = bf[tid]; BLs[1][tid] = bi1[tid];
        BLs[2][tid] = bi2[tid]; BLs[3][tid] = bo[tid];
    }
    if (tid < 48) {
        float s = bn1_g[tid] * rsqrtf(bn1_v[tid] + 1e-3f);
        sL[tid] = s;
        tL[tid] = bn1_b[tid] - bn1_m[tid] * s;
    }
    const int b = tid >> 3, u = tid & 7;
    hL[b][u] = 0.f;
    float c = 0.f;

    for (int t = 0; t < 32; ++t) {
        for (int i = tid; i < 32 * 48; i += 256) {
            int bb = i / 48, cc = i - (i / 48) * 48;
            featL[bb][cc] = sL[cc] * (featsum[(bb * 32 + t) * 48 + cc] * (1.f / 784.f)) + tL[cc];
        }
        __syncthreads();

        float zf = BLs[0][u], zi = BLs[1][u], zg = BLs[2][u], zo = BLs[3][u];
#pragma unroll 8
        for (int j = 0; j < 48; ++j) {
            float v = featL[b][j];
            zf += v * WL[0][j][u]; zi += v * WL[1][j][u];
            zg += v * WL[2][j][u]; zo += v * WL[3][j][u];
        }
#pragma unroll
        for (int j = 0; j < 8; ++j) {
            float v = hL[b][j];
            zf += v * WL[0][48 + j][u]; zi += v * WL[1][48 + j][u];
            zg += v * WL[2][48 + j][u]; zo += v * WL[3][48 + j][u];
        }
        __syncthreads();

        float fg = 1.f / (1.f + __expf(-zf));
        float ig = 1.f / (1.f + __expf(-zi));
        float gg = tanhf(zg);
        float og = 1.f / (1.f + __expf(-zo));
        c = fg * c + ig * gg;
        float h = og * tanhf(c);
        hL[b][u] = h;
        __syncthreads();

        if (u == 0) {
            float y = b_out[0];
#pragma unroll
            for (int j = 0; j < 8; ++j) {
                float s2 = bn2_g[j] * rsqrtf(bn2_v[j] + 1e-3f);
                float hv = hL[b][j];
                y += ((hv - bn2_m[j]) * s2 + bn2_b[j]) * w_out[j];
            }
            out[b * 32 + t] = y;
        }
    }
}

// ---------------- launch ----------------
extern "C" void kernel_launch(void* const* d_in, const int* in_sizes, int n_in,
                              void* d_out, int out_size, void* d_ws, size_t ws_size,
                              hipStream_t stream) {
    (void)in_sizes; (void)n_in; (void)out_size;
    if (ws_size < WS_NEEDED) return;

    const float* x     = (const float*)d_in[0];
    const float* w1    = (const float*)d_in[1];
    const float* b1    = (const float*)d_in[2];
    const float* w2    = (const float*)d_in[3];
    const float* b2    = (const float*)d_in[4];
    const float* w3    = (const float*)d_in[5];
    const float* b3    = (const float*)d_in[6];
    const float* bn1_g = (const float*)d_in[7];
    const float* bn1_b = (const float*)d_in[8];
    const float* bn1_m = (const float*)d_in[9];
    const float* bn1_v = (const float*)d_in[10];
    const float* wf    = (const float*)d_in[11];
    const float* bf    = (const float*)d_in[12];
    const float* wi1   = (const float*)d_in[13];
    const float* bi1   = (const float*)d_in[14];
    const float* wi2   = (const float*)d_in[15];
    const float* bi2   = (const float*)d_in[16];
    const float* wo    = (const float*)d_in[17];
    const float* bo    = (const float*)d_in[18];
    const float* bn2_g = (const float*)d_in[19];
    const float* bn2_b = (const float*)d_in[20];
    const float* bn2_m = (const float*)d_in[21];
    const float* bn2_v = (const float*)d_in[22];
    const float* w_out = (const float*)d_in[23];
    const float* b_out = (const float*)d_in[24];

    char* ws = (char*)d_ws;
    _Float16* act1    = (_Float16*)(ws + OFF_ACT1);
    _Float16* act2    = (_Float16*)(ws + OFF_ACT2);
    float*    featsum = (float*)   (ws + OFF_FEAT);
    _Float16* Bt2     = (_Float16*)(ws + OFF_BT2);
    _Float16* Bt3     = (_Float16*)(ws + OFF_BT3);

    zero_ws_kernel<<<4096, 256, 0, stream>>>((uint4*)ws, (int)(ZERO_BYTES / 16));
    prep_weights_kernel<<<(48 * 448 + 255) / 256, 256, 0, stream>>>(w2, w3, Bt2, Bt3);
    conv1_kernel<<<3136, 256, 0, stream>>>(x, w1, b1, act1);
    conv48_wmma_kernel<false><<<6272, 256, 0, stream>>>(act1, Bt2, b2, act2, nullptr);
    conv48_wmma_kernel<true ><<<6272, 256, 0, stream>>>(act2, Bt3, b3, nullptr, featsum);
    lstm_kernel<<<1, 256, 0, stream>>>(featsum, bn1_g, bn1_b, bn1_m, bn1_v,
                                       wf, bf, wi1, bi1, wi2, bi2, wo, bo,
                                       bn2_g, bn2_b, bn2_m, bn2_v, w_out, b_out,
                                       (float*)d_out);
}
